// MRN4GRUCell_59390807769125
// MI455X (gfx1250) — matrix operand order repaired and checked by reading, not compiled, not run on hardware
//
#include <hip/hip_runtime.h>

// ---------------------------------------------------------------------------
// MRN4GRUCell fused, MI455X (gfx1250, wave32, WMMA)
//
// Roofline: required compute (selected-type GRU only) = ~13 GFLOP;
// required memory = ~0.67 GB -> ~29 us @ 23.3 TB/s. Memory bound -> fuse
// everything into one pass; use f16 WMMA for the GEMM so compute stays
// far under the memory roofline.
//
// Pipeline (all on `stream`):
//   1) prep    : W_ih/W_hh (T,3H,K) f32 -> Wt (K, T*3H) f16 in ws; zero counts
//   2) bucket  : rows grouped by type (LDS-aggregated atomics)
//   3) tsetup  : per-type 32-row tile offsets (prefix sum, 1 thread)
//   4) fused   : per wave: one 32-row tile of one type
//        - A operands: input & c rows -> f16, WMMA A layout
//        - 8 h-chunks x (6 gate tiles x 2 row tiles x 4 K-steps) WMMAs
//        - gate math in f32 (sigmoid/tanh via v_exp_f32), hidden -> LDS
//        - attention: permute {hidden, reg[...]}, shfl-reduce dots,
//          softmax over T=8, write context + reg_new
// ---------------------------------------------------------------------------

#define BB   65536   // batch
#define HH   128     // hidden
#define II   128     // input size (== K for both GEMMs)
#define TN   8       // types
#define G3   384     // 3*H
#define NW   3072    // T*3H
#define ROWS 32      // rows per wave (2x 16-row WMMA tiles)
#define WPB  2       // waves per block

// workspace layout (bytes)
#define WS_WTIH 0
#define WS_WTHH (WS_WTIH + NW * II * 2)
#define WS_CNT  (WS_WTHH + NW * II * 2)
#define WS_TB   (WS_CNT + 32)
#define WS_LIST (WS_CNT + 128)

typedef __attribute__((ext_vector_type(16))) _Float16 v16h;
typedef __attribute__((ext_vector_type(8)))  float    v8f;

#define WMMA_F16(a, b, acc) \
  __builtin_amdgcn_wmma_f32_16x16x32_f16(false, (a), false, (b), (short)0, (acc), false, false)

__device__ __forceinline__ float sigm_f(float x) { return 1.0f / (1.0f + __expf(-x)); }
__device__ __forceinline__ float tanh_f(float x) {
  float e = __expf(2.0f * x);
  return 1.0f - 2.0f / (e + 1.0f);
}

// Build WMMA A-operand v16h for one lane from a row-major f32 row.
// Per ISA A layout (16-bit 16x32): lanes 0-15 row M, halves = K {k0..k0+7,
// k0+16..k0+23}; lanes 16-31 same M rows, k0 += 8.
__device__ __forceinline__ v16h load_a_f32(const float* __restrict__ rowp, int k0) {
  const float4* p = (const float4*)(rowp + k0);
  float4 x0 = p[0], x1 = p[1];   // K = k0 .. k0+7
  float4 y0 = p[4], y1 = p[5];   // K = k0+16 .. k0+23
  v16h a;
  a[0]  = (_Float16)x0.x; a[1]  = (_Float16)x0.y; a[2]  = (_Float16)x0.z; a[3]  = (_Float16)x0.w;
  a[4]  = (_Float16)x1.x; a[5]  = (_Float16)x1.y; a[6]  = (_Float16)x1.z; a[7]  = (_Float16)x1.w;
  a[8]  = (_Float16)y0.x; a[9]  = (_Float16)y0.y; a[10] = (_Float16)y0.z; a[11] = (_Float16)y0.w;
  a[12] = (_Float16)y1.x; a[13] = (_Float16)y1.y; a[14] = (_Float16)y1.z; a[15] = (_Float16)y1.w;
  return a;
}

// ---------------------------------------------------------------------------
// 1) weights f32 (N=T*3H rows, K cols) -> f16 transposed Wt[K][N]; zero counts
__global__ void mrn_prep(const float* __restrict__ Wih, const float* __restrict__ Whh,
                         _Float16* __restrict__ wtih, _Float16* __restrict__ wthh,
                         int* __restrict__ cnt) {
  if (blockIdx.x == 0 && threadIdx.x < TN) cnt[threadIdx.x] = 0;
  int e = blockIdx.x * 256 + threadIdx.x;          // 0 .. 2*NW*II-1
  const float* src = (e < NW * II) ? Wih : Whh;
  _Float16*    dst = (e < NW * II) ? wtih : wthh;
  int e2 = (e < NW * II) ? e : (e - NW * II);
  int k = e2 / NW;
  int n = e2 % NW;
  dst[e2] = (_Float16)src[n * II + k];             // Wt[k][n] = W[n][k]
}

// ---------------------------------------------------------------------------
// 2) bucket rows by type (LDS-aggregated to keep global atomics ~2K total)
__global__ void mrn_bucket(const int* __restrict__ type, int* __restrict__ cnt,
                           int* __restrict__ list) {
  __shared__ int lcnt[TN];
  __shared__ int lbase[TN];
  int tid = threadIdx.x;
  if (tid < TN) lcnt[tid] = 0;
  __syncthreads();
  int b = blockIdx.x * 256 + tid;
  int t = type[b];
  int lp = atomicAdd(&lcnt[t], 1);
  __syncthreads();
  if (tid < TN) lbase[tid] = atomicAdd(&cnt[tid], lcnt[tid]);
  __syncthreads();
  list[t * BB + lbase[t] + lp] = b;
}

// ---------------------------------------------------------------------------
// 3) per-type tile bases (prefix sum over ceil(cnt/32))
__global__ void mrn_tsetup(const int* __restrict__ cnt, int* __restrict__ tb) {
  if (threadIdx.x == 0 && blockIdx.x == 0) {
    int cum = 0;
    for (int t = 0; t < TN; ++t) {
      tb[t] = cum;
      cum += (cnt[t] + ROWS - 1) / ROWS;
    }
    tb[TN] = cum;
  }
}

// ---------------------------------------------------------------------------
// 4) fused GRU (selected type only, via WMMA) + permutation + attention
__global__ __launch_bounds__(WPB * 32) void mrn_fused(
    const float* __restrict__ query, const float* __restrict__ input,
    const float* __restrict__ c, const float* __restrict__ reg,
    const float* __restrict__ b_ih, const float* __restrict__ b_hh,
    const _Float16* __restrict__ wtih, const _Float16* __restrict__ wthh,
    const int* __restrict__ cnt, const int* __restrict__ tb,
    const int* __restrict__ list, float* __restrict__ out) {
  __shared__ int   rowsS[WPB][ROWS];
  __shared__ float htS[WPB][ROWS][132];            // pad 132 to dodge bank conflicts

  const int lane = threadIdx.x & 31;
  const int wv   = threadIdx.x >> 5;
  const int g    = blockIdx.x * WPB + wv;          // global wave/tile id
  const int total = tb[TN];
  if (g >= total) return;                          // wave-uniform exit

  int t = 0;
  while (t < TN - 1 && g >= tb[t + 1]) ++t;
  const int tileIdx = g - tb[t];
  const int cn = cnt[t];

  // gather this tile's 32 (possibly clamped-duplicate) batch rows
  {
    int j = tileIdx * ROWS + lane;
    if (j >= cn) j = cn - 1;
    rowsS[wv][lane] = list[t * BB + j];
  }

  // ---- A operands: input and c rows as f16 WMMA A tiles (K = 128 = 4x32) ---
  v16h a_in[2][4], a_c[2][4];
  const int m16 = lane & 15;
  const int hiK = (lane >> 4) * 8;
#pragma unroll
  for (int rt = 0; rt < 2; ++rt) {
    int row = rowsS[wv][rt * 16 + m16];
    const float* pi = input + (size_t)row * II;
    const float* pc = c + (size_t)row * HH;
#pragma unroll
    for (int kc = 0; kc < 4; ++kc) {
      a_in[rt][kc] = load_a_f32(pi, kc * 32 + hiK);
      a_c[rt][kc]  = load_a_f32(pc, kc * 32 + hiK);
    }
  }

  const int nlo   = lane & 15;                     // N within 16-wide tile
  const int mbase = (lane >> 4) << 3;              // +8 rows for lanes 16-31

  // ---- GRU gates per 16-wide h-chunk --------------------------------------
  const v8f vz = {0.f, 0.f, 0.f, 0.f, 0.f, 0.f, 0.f, 0.f};
#pragma unroll 1
  for (int hc = 0; hc < 8; ++hc) {
    const int h0 = hc * 16;
    const int nr = t * G3 + h0;                    // r-gate column base
    const int nz = nr + HH;                        // z-gate
    const int nn = nr + 2 * HH;                    // n-gate
    v8f ir[2] = {vz, vz}, iz[2] = {vz, vz}, inn[2] = {vz, vz};
    v8f hr[2] = {vz, vz}, hz[2] = {vz, vz}, hnn[2] = {vz, vz};

#pragma unroll
    for (int kc = 0; kc < 4; ++kc) {
      // B operand: lane%16 = N column, v16h = K {hiK..hiK+7, hiK+16..hiK+23}
      const _Float16* wi = wtih + (size_t)(kc * 32 + hiK) * NW + nlo;
      const _Float16* wh = wthh + (size_t)(kc * 32 + hiK) * NW + nlo;
      v16h bir, biz, bin, bhr, bhz, bhn;
#pragma unroll
      for (int s = 0; s < 16; ++s) {
        size_t ko = (size_t)(s < 8 ? s : s + 8) * NW;
        bir[s] = wi[ko + nr]; biz[s] = wi[ko + nz]; bin[s] = wi[ko + nn];
        bhr[s] = wh[ko + nr]; bhz[s] = wh[ko + nz]; bhn[s] = wh[ko + nn];
      }
#pragma unroll
      for (int rt = 0; rt < 2; ++rt) {
        ir[rt]  = WMMA_F16(a_in[rt][kc], bir, ir[rt]);
        iz[rt]  = WMMA_F16(a_in[rt][kc], biz, iz[rt]);
        inn[rt] = WMMA_F16(a_in[rt][kc], bin, inn[rt]);
        hr[rt]  = WMMA_F16(a_c[rt][kc], bhr, hr[rt]);
        hz[rt]  = WMMA_F16(a_c[rt][kc], bhz, hz[rt]);
        hnn[rt] = WMMA_F16(a_c[rt][kc], bhn, hnn[rt]);
      }
    }

    const float bir = b_ih[nr + nlo], biz = b_ih[nz + nlo], bin = b_ih[nn + nlo];
    const float bhr = b_hh[nr + nlo], bhz = b_hh[nz + nlo], bhn = b_hh[nn + nlo];

#pragma unroll
    for (int rt = 0; rt < 2; ++rt) {
#pragma unroll
      for (int i = 0; i < 8; ++i) {
        int m = rt * 16 + mbase + i;               // C/D layout: VGPR i -> row
        int row = rowsS[wv][m];
        float cv = c[(size_t)row * HH + h0 + nlo];
        float R  = sigm_f(ir[rt][i] + bir + hr[rt][i] + bhr);
        float Z  = sigm_f(iz[rt][i] + biz + hz[rt][i] + bhz);
        float Nv = tanh_f(inn[rt][i] + bin + R * (hnn[rt][i] + bhn));
        htS[wv][m][h0 + nlo] = (1.0f - Z) * Nv + Z * cv;
      }
    }
  }

  // ---- permutation + attention (torch-faithful index quirk) ---------------
  const float inv_sqrt_h = 0.08838834764831845f;   // 1/sqrt(128)
  const int kh = (t == 0) ? 0 : 1;                 // slot that receives hidden
  float* out_ctx = out;
  float* out_reg = out + (size_t)BB * HH;
  const int h = lane * 4;                          // 32 lanes x float4 = 128

#pragma unroll 1
  for (int r = 0; r < ROWS; ++r) {
    int row = rowsS[wv][r];
    float4 q = *(const float4*)(query + (size_t)row * HH + h);
    float4 v[TN];
    float  sc[TN];
#pragma unroll
    for (int k = 0; k < TN; ++k) {
      float4 vv;
      if (k == kh) {
        vv = *(const float4*)(&htS[wv][r][h]);
      } else {
        int src = (t == 0) ? k : (k == 0 ? (t - 1) : (k <= t ? (k - 2) : k));
        vv = *(const float4*)(reg + (size_t)row * TN * HH + (size_t)src * HH + h);
      }
      v[k]  = vv;
      sc[k] = vv.x * q.x + vv.y * q.y + vv.z * q.z + vv.w * q.w;
      *(float4*)(out_reg + (size_t)row * TN * HH + (size_t)k * HH + h) = vv;
    }
#pragma unroll
    for (int k = 0; k < TN; ++k) {
      float s = sc[k];
      for (int off = 16; off > 0; off >>= 1) s += __shfl_xor(s, off, 32);
      sc[k] = s * inv_sqrt_h;
    }
    float mx = sc[0];
#pragma unroll
    for (int k = 1; k < TN; ++k) mx = fmaxf(mx, sc[k]);
    float w[TN], sum = 0.f;
#pragma unroll
    for (int k = 0; k < TN; ++k) { w[k] = __expf(sc[k] - mx); sum += w[k]; }
    float inv = 1.0f / sum;
    float4 ctx = {0.f, 0.f, 0.f, 0.f};
#pragma unroll
    for (int k = 0; k < TN; ++k) {
      float wk = w[k] * inv;
      ctx.x += wk * v[k].x; ctx.y += wk * v[k].y;
      ctx.z += wk * v[k].z; ctx.w += wk * v[k].w;
    }
    *(float4*)(out_ctx + (size_t)row * HH + h) = ctx;
  }
}

// ---------------------------------------------------------------------------
extern "C" void kernel_launch(void* const* d_in, const int* in_sizes, int n_in,
                              void* d_out, int out_size, void* d_ws, size_t ws_size,
                              hipStream_t stream) {
  (void)in_sizes; (void)n_in; (void)out_size; (void)ws_size;
  const float* query = (const float*)d_in[0];
  const float* input = (const float*)d_in[1];
  const int*   type  = (const int*)d_in[2];
  const float* c     = (const float*)d_in[3];
  const float* reg   = (const float*)d_in[4];
  const float* W_ih  = (const float*)d_in[5];
  const float* W_hh  = (const float*)d_in[6];
  const float* b_ih  = (const float*)d_in[7];
  const float* b_hh  = (const float*)d_in[8];
  float* out = (float*)d_out;

  char* ws = (char*)d_ws;
  _Float16* wtih = (_Float16*)(ws + WS_WTIH);
  _Float16* wthh = (_Float16*)(ws + WS_WTHH);
  int* cntp  = (int*)(ws + WS_CNT);
  int* tbp   = (int*)(ws + WS_TB);
  int* listp = (int*)(ws + WS_LIST);

  mrn_prep<<<(2 * NW * II) / 256, 256, 0, stream>>>(W_ih, W_hh, wtih, wthh, cntp);
  mrn_bucket<<<BB / 256, 256, 0, stream>>>(type, cntp, listp);
  mrn_tsetup<<<1, 32, 0, stream>>>(cntp, tbp);

  const int maxTiles = BB / ROWS + TN;             // >= sum ceil(cnt_t/32)
  const int blocks = (maxTiles + WPB - 1) / WPB;
  mrn_fused<<<blocks, WPB * 32, 0, stream>>>(query, input, c, reg, b_ih, b_hh,
                                             wtih, wthh, cntp, tbp, listp, out);
}
